// CustomQwen3Attention_44152263803384
// MI455X (gfx1250) — matrix operand order repaired and checked by reading, not compiled, not run on hardware
//
#include <hip/hip_runtime.h>

// ---------------------------------------------------------------------------
// Qwen3-style attention block for MI455X (gfx1250), bf16 WMMA pipeline.
// Round 4: fix async-to-LDS builtin arg types (expects v4i pointers per the
// compiler diagnostic). Double-buffered LDS with ASYNCcnt-tracked DMA staging.
// ---------------------------------------------------------------------------

typedef __bf16 bf16;
typedef __attribute__((ext_vector_type(8)))  __bf16 v8bf;
typedef __attribute__((ext_vector_type(16))) __bf16 v16bf;
typedef __attribute__((ext_vector_type(8)))  float  v8f;
typedef __attribute__((__vector_size__(4 * sizeof(int)))) int v4i;

static constexpr int SEQ   = 2048;   // q_len
static constexpr int HIDN  = 2048;   // hidden size
static constexpr int NHQ   = 16;     // query heads
static constexpr int NKVH  = 8;      // kv heads
static constexpr int HD    = 128;    // head dim
static constexpr int PASTL = 2048;   // past length
static constexpr int KVTOT = PASTL + SEQ; // 4096
static constexpr float EPSV = 1e-6f;

#if defined(__HIP_DEVICE_COMPILE__) && __has_builtin(__builtin_amdgcn_global_load_async_to_lds_b128)
#define HAVE_ASYNC_LDS 1
#else
#define HAVE_ASYNC_LDS 0
#endif

#define AS1 __attribute__((address_space(1)))
#define AS3 __attribute__((address_space(3)))

#if HAVE_ASYNC_LDS
// 16-byte async copy global -> LDS (tracked by ASYNCcnt, data bypasses VGPRs)
__device__ __forceinline__ void async_cp16(const bf16* g, bf16* l) {
  __builtin_amdgcn_global_load_async_to_lds_b128(
      (AS1 v4i*)(const_cast<bf16*>(g)), (AS3 v4i*)l, 0, 0);
}
#endif

__device__ __forceinline__ void wait_async0() {
#if defined(__HIP_DEVICE_COMPILE__)
#if __has_builtin(__builtin_amdgcn_s_wait_asynccnt)
  __builtin_amdgcn_s_wait_asynccnt(0);
#else
  asm volatile("s_wait_asynccnt 0x0" ::: "memory");
#endif
#endif
}

__device__ __forceinline__ v16bf pack2(v8bf lo, v8bf hi) {
  v16bf r;
#pragma unroll
  for (int i = 0; i < 8; ++i) { r[i] = lo[i]; r[i + 8] = hi[i]; }
  return r;
}

// A fragment (16x32 bf16, MxK): lane m holds K = half*8..+7 (VGPR0-3) and
// K = 16+half*8..+7 (VGPR4-7). Caller passes pointer already offset by half*8.
__device__ __forceinline__ v16bf load_a_frag(const bf16* p) {
  return pack2(*(const v8bf*)p, *(const v8bf*)(p + 16));
}
// B fragment (32x16 bf16, KxN): lane holds column n, K = half*8..+7 and
// K = 16+half*8..+7. Caller passes pointer already offset by half*8.
__device__ __forceinline__ v16bf load_b_frag(const bf16* p) {
  return pack2(*(const v8bf*)p, *(const v8bf*)(p + 16));
}

// ---------------------------------------------------------------------------
// fp32 -> bf16 conversions
// ---------------------------------------------------------------------------
__global__ void f32_to_bf16_kernel(const float* __restrict__ src,
                                   bf16* __restrict__ dst, int n) {
  int i = blockIdx.x * blockDim.x + threadIdx.x;
  if (i < n) dst[i] = (bf16)src[i];
}

__global__ void past_key_kernel(const float* __restrict__ src,
                                bf16* __restrict__ dst) {
  int i = blockIdx.x * blockDim.x + threadIdx.x;
  if (i >= NKVH * PASTL * HD) return;
  int h = i / (PASTL * HD);
  int r = i % (PASTL * HD);
  dst[(size_t)h * KVTOT * HD + r] = (bf16)src[i];
}

__global__ void past_value_kernel(const float* __restrict__ src,
                                  bf16* __restrict__ dst) {
  int i = blockIdx.x * blockDim.x + threadIdx.x;
  if (i >= NKVH * PASTL * HD) return;
  int h   = i / (PASTL * HD);
  int rem = i % (PASTL * HD);
  int kv  = rem / HD;
  int d   = rem % HD;
  dst[((size_t)h * HD + d) * KVTOT + kv] = (bf16)src[i];
}

__global__ void v_finish_kernel(const float* __restrict__ vraw,
                                float* __restrict__ new_value,
                                bf16* __restrict__ vt) {
  int i = blockIdx.x * blockDim.x + threadIdx.x;
  if (i >= SEQ * NKVH * HD) return;
  int s   = i / (NKVH * HD);
  int rem = i % (NKVH * HD);
  int h   = rem / HD;
  int d   = rem % HD;
  float val = vraw[i];
  new_value[((size_t)h * SEQ + s) * HD + d] = val;
  vt[((size_t)h * HD + d) * KVTOT + (PASTL + s)] = (bf16)val;
}

// ---------------------------------------------------------------------------
// RMSNorm + RoPE. raw: [SEQ][nheads*HD] fp32. One block = one (s, head) row.
// ---------------------------------------------------------------------------
__global__ void __launch_bounds__(128)
rmsnorm_rope_kernel(const float* __restrict__ raw,
                    const float* __restrict__ cosb, const float* __restrict__ sinb,
                    const float* __restrict__ w,
                    float* __restrict__ out_f32, bf16* __restrict__ out_bf16,
                    int nheads, float scale, int kv_stride, int kv_offset) {
  __shared__ float red[HD];
  __shared__ float xnorm[HD];
  int t    = threadIdx.x;
  int sIdx = blockIdx.x / nheads;
  int hh   = blockIdx.x % nheads;
  float x = raw[((size_t)sIdx * nheads + hh) * HD + t];
  red[t] = x * x;
  __syncthreads();
#pragma unroll
  for (int off = 64; off > 0; off >>= 1) {
    if (t < off) red[t] += red[t + off];
    __syncthreads();
  }
  float inv = rsqrtf(red[0] * (1.0f / HD) + EPSV);
  float xn  = x * inv * w[t];
  xnorm[t] = xn;
  __syncthreads();
  float rh = (t < HD / 2) ? -xnorm[t + HD / 2] : xnorm[t - HD / 2];
  float cs = cosb[(size_t)sIdx * HD + t];
  float sn = sinb[(size_t)sIdx * HD + t];
  float o  = xn * cs + rh * sn;
  if (out_f32) out_f32[((size_t)hh * SEQ + sIdx) * HD + t] = o;
  out_bf16[((size_t)hh * kv_stride + kv_offset + sIdx) * HD + t] = (bf16)(o * scale);
}

// ---------------------------------------------------------------------------
// Generic bf16 GEMM: C[M,N] fp32 = A[M,K] * B[N,K]^T   (K contiguous in both)
// 256 threads (8 waves). Tile 128x128, K-tile 64; wave = 32x64 (2x4 tiles).
// Double-buffered LDS (flat array + buf*BUFSZ offset: stays addrspace(3)).
// ---------------------------------------------------------------------------
static constexpr int GBM = 128, GBN = 128, GBK = 64;
static constexpr int GSTR = GBK + 8;          // bf16 elems, keeps 16B alignment
static constexpr int ABUF = GBM * GSTR;
static constexpr int BBUF = GBN * GSTR;

__global__ void __launch_bounds__(256)
gemm_bf16_kernel(const bf16* __restrict__ A, const bf16* __restrict__ B,
                 float* __restrict__ C, int M, int N, int K) {
  __shared__ bf16 Alds[2 * ABUF];
  __shared__ bf16 Blds[2 * BBUF];
  int tid = threadIdx.x;
  int lane = tid & 31, wid = tid >> 5;
  int wm = wid & 3, wn = wid >> 2;
  int n15 = lane & 15, half = lane >> 4;
  int M0 = blockIdx.y * GBM, N0 = blockIdx.x * GBN;

  // loader mapping: each thread stages 32 elems (4x 16B) of A and of B
  int lr = tid >> 1, lc = (tid & 1) * 32;
  const bf16* Agp = A + (size_t)(M0 + lr) * K + lc;
  const bf16* Bgp = B + (size_t)(N0 + lr) * K + lc;
  int ldsOff = lr * GSTR + lc;

  v8f acc[2][4];
#pragma unroll
  for (int mt = 0; mt < 2; ++mt)
#pragma unroll
    for (int nt = 0; nt < 4; ++nt) acc[mt][nt] = (v8f)(0.0f);

#if HAVE_ASYNC_LDS
  // prologue: async-stage K-tile 0 into buffer 0
#pragma unroll
  for (int j = 0; j < 4; ++j) {
    async_cp16(Agp + j * 8, Alds + ldsOff + j * 8);
    async_cp16(Bgp + j * 8, Blds + ldsOff + j * 8);
  }
  wait_async0();
  __syncthreads();
#else
  uint4 ar[4], br[4];
#pragma unroll
  for (int j = 0; j < 4; ++j) {
    ar[j] = ((const uint4*)Agp)[j];
    br[j] = ((const uint4*)Bgp)[j];
  }
#pragma unroll
  for (int j = 0; j < 4; ++j) {
    ((uint4*)(Alds + ldsOff))[j] = ar[j];
    ((uint4*)(Blds + ldsOff))[j] = br[j];
  }
  __syncthreads();
#endif

  int buf = 0;
  for (int k0 = 0; k0 < K; k0 += GBK) {
    bool hasNext = (k0 + GBK) < K;
#if HAVE_ASYNC_LDS
    if (hasNext) {  // async DMA of next tile overlaps with WMMA below
      int nb = buf ^ 1;
#pragma unroll
      for (int j = 0; j < 4; ++j) {
        async_cp16(Agp + k0 + GBK + j * 8, Alds + nb * ABUF + ldsOff + j * 8);
        async_cp16(Bgp + k0 + GBK + j * 8, Blds + nb * BBUF + ldsOff + j * 8);
      }
    }
#else
    if (hasNext) {
      const uint4* ag = (const uint4*)(Agp + k0 + GBK);
      const uint4* bg = (const uint4*)(Bgp + k0 + GBK);
#pragma unroll
      for (int j = 0; j < 4; ++j) { ar[j] = ag[j]; br[j] = bg[j]; }
    }
#endif
    const bf16* Ab = Alds + buf * ABUF;
    const bf16* Bb = Blds + buf * BBUF;
#pragma unroll
    for (int kk = 0; kk < GBK; kk += 32) {
      v16bf af[2], bfr[4];
#pragma unroll
      for (int mt = 0; mt < 2; ++mt)
        af[mt] = load_a_frag(Ab + (wm * 32 + mt * 16 + n15) * GSTR + kk + half * 8);
#pragma unroll
      for (int nt = 0; nt < 4; ++nt)
        bfr[nt] = load_b_frag(Bb + (wn * 64 + nt * 16 + n15) * GSTR + kk + half * 8);
#pragma unroll
      for (int mt = 0; mt < 2; ++mt)
#pragma unroll
        for (int nt = 0; nt < 4; ++nt)
          acc[mt][nt] = __builtin_amdgcn_wmma_f32_16x16x32_bf16(
              false, af[mt], false, bfr[nt], (short)0, acc[mt][nt], false, false);
    }
#if HAVE_ASYNC_LDS
    if (hasNext) wait_async0();
#else
    if (hasNext) {
      int nb = buf ^ 1;
#pragma unroll
      for (int j = 0; j < 4; ++j) {
        ((uint4*)(Alds + nb * ABUF + ldsOff))[j] = ar[j];
        ((uint4*)(Blds + nb * BBUF + ldsOff))[j] = br[j];
      }
    }
#endif
    __syncthreads();
    buf ^= 1;
  }

#pragma unroll
  for (int mt = 0; mt < 2; ++mt) {
#pragma unroll
    for (int nt = 0; nt < 4; ++nt) {
#pragma unroll
      for (int i = 0; i < 8; ++i) {
        int r = M0 + wm * 32 + mt * 16 + i + 8 * half;
        int c = N0 + wn * 64 + nt * 16 + n15;
        C[(size_t)r * N + c] = acc[mt][nt][i];
      }
    }
  }
}

// ---------------------------------------------------------------------------
// Flash attention (causal, GQA 2:1).
// Q: [NHQ][SEQ][HD] bf16 (pre-scaled by HD^-0.5)
// K: [NKVH][KVTOT][HD] bf16     V: [NKVH][HD][KVTOT] bf16 (transposed)
// Out: [SEQ][NHQ*HD] bf16
// 256 threads, 128 q-rows/block, wave = 16 rows; KV chunk 64, double-buffered.
// ---------------------------------------------------------------------------
static constexpr int ATM = 128, CK = 64;
static constexpr int KSTR = HD + 8;    // 136
static constexpr int VSTR = CK + 8;    // 72
static constexpr int PSTR = CK + 8;    // 72
static constexpr int KBUF = CK * KSTR;
static constexpr int VBUF = HD * VSTR;

__global__ void __launch_bounds__(256)
attn_kernel(const bf16* __restrict__ Qm, const bf16* __restrict__ Km,
            const bf16* __restrict__ Vt, bf16* __restrict__ Om) {
  __shared__ bf16 Klds[2 * KBUF];
  __shared__ bf16 Vlds[2 * VBUF];
  __shared__ bf16 Plds[8 * 16 * PSTR];

  int tid = threadIdx.x, lane = tid & 31, wid = tid >> 5;
  int n15 = lane & 15, half = lane >> 4;
  int h = blockIdx.x, kvh = h >> 1;
  int q0 = blockIdx.y * ATM;
  int wrow0 = q0 + wid * 16;

  // Q fragments for this wave's 16 rows (full HD=128 => 4 x K=32)
  v16bf aq[4];
  const bf16* qbase = Qm + ((size_t)h * SEQ + wrow0 + n15) * HD;
#pragma unroll
  for (int kf = 0; kf < 4; ++kf)
    aq[kf] = load_a_frag(qbase + kf * 32 + half * 8);

  v8f o[8];
#pragma unroll
  for (int i = 0; i < 8; ++i) o[i] = (v8f)(0.0f);
  float mrow[8], lrow[8];
#pragma unroll
  for (int i = 0; i < 8; ++i) { mrow[i] = -3.0e38f; lrow[i] = 0.0f; }

  const bf16* Kb = Km + (size_t)kvh * KVTOT * HD;
  const bf16* Vb = Vt + (size_t)kvh * HD * KVTOT;
  bf16* pb = Plds + wid * 16 * PSTR;

  // loader mapping
  int kr = tid >> 2, kc = (tid & 3) * 32;     // K tile: 64 rows x 128 hd
  int vr = tid >> 1, vc = (tid & 1) * 32;     // V tile: 128 hd rows x 64 kv
  const bf16* Kgp = Kb + (size_t)kr * HD + kc;
  const bf16* Vgp = Vb + (size_t)vr * KVTOT + vc;
  int kOff = kr * KSTR + kc;
  int vOff = vr * VSTR + vc;

#if HAVE_ASYNC_LDS
#pragma unroll
  for (int j = 0; j < 4; ++j) {
    async_cp16(Kgp + j * 8, Klds + kOff + j * 8);
    async_cp16(Vgp + j * 8, Vlds + vOff + j * 8);
  }
  wait_async0();
  __syncthreads();
#else
  uint4 krg[4], vrg[4];
#pragma unroll
  for (int j = 0; j < 4; ++j) {
    krg[j] = ((const uint4*)Kgp)[j];
    vrg[j] = ((const uint4*)Vgp)[j];
  }
#pragma unroll
  for (int j = 0; j < 4; ++j) {
    ((uint4*)(Klds + kOff))[j] = krg[j];
    ((uint4*)(Vlds + vOff))[j] = vrg[j];
  }
  __syncthreads();
#endif

  int buf = 0;
  int kvmax = PASTL + q0 + ATM;  // exclusive causal bound for this block
  for (int c0 = 0; c0 < kvmax; c0 += CK) {
    bool hasNext = (c0 + CK) < kvmax;
#if HAVE_ASYNC_LDS
    if (hasNext) {
      int nb = buf ^ 1;
#pragma unroll
      for (int j = 0; j < 4; ++j) {
        async_cp16(Kgp + (size_t)(c0 + CK) * HD + j * 8, Klds + nb * KBUF + kOff + j * 8);
        async_cp16(Vgp + (c0 + CK) + j * 8, Vlds + nb * VBUF + vOff + j * 8);
      }
    }
#else
    if (hasNext) {
      const uint4* kg = (const uint4*)(Kgp + (size_t)(c0 + CK) * HD);
      const uint4* vg = (const uint4*)(Vgp + (c0 + CK));
#pragma unroll
      for (int j = 0; j < 4; ++j) { krg[j] = kg[j]; vrg[j] = vg[j]; }
    }
#endif
    const bf16* Kc = Klds + buf * KBUF;
    const bf16* Vc = Vlds + buf * VBUF;

    // S = Q K^T over this chunk: 4 tiles of 16x16, contraction over HD
    v8f st[4];
#pragma unroll
    for (int t = 0; t < 4; ++t) st[t] = (v8f)(0.0f);
#pragma unroll
    for (int kf = 0; kf < 4; ++kf) {
      v16bf bfr[4];
#pragma unroll
      for (int t = 0; t < 4; ++t)
        bfr[t] = load_b_frag(Kc + (t * 16 + n15) * KSTR + kf * 32 + half * 8);
#pragma unroll
      for (int t = 0; t < 4; ++t)
        st[t] = __builtin_amdgcn_wmma_f32_16x16x32_bf16(
            false, aq[kf], false, bfr[t], (short)0, st[t], false, false);
    }

    // causal mask
#pragma unroll
    for (int t = 0; t < 4; ++t) {
      int col = c0 + t * 16 + n15;
#pragma unroll
      for (int i = 0; i < 8; ++i) {
        int row = wrow0 + i + 8 * half;
        if (col > PASTL + row) st[t][i] = -1.0e9f;
      }
    }

    // online softmax (row = i + 8*half, striped across 16 lanes)
    float newm[8], alpha[8], psum[8];
#pragma unroll
    for (int i = 0; i < 8; ++i) {
      float mx = fmaxf(fmaxf(st[0][i], st[1][i]), fmaxf(st[2][i], st[3][i]));
      mx = fmaxf(mx, __shfl_xor(mx, 1, 32));
      mx = fmaxf(mx, __shfl_xor(mx, 2, 32));
      mx = fmaxf(mx, __shfl_xor(mx, 4, 32));
      mx = fmaxf(mx, __shfl_xor(mx, 8, 32));
      newm[i]  = fmaxf(mrow[i], mx);
      alpha[i] = __expf(mrow[i] - newm[i]);
      mrow[i]  = newm[i];
      psum[i]  = 0.0f;
    }
#pragma unroll
    for (int t = 0; t < 4; ++t) {
#pragma unroll
      for (int i = 0; i < 8; ++i) {
        float p = __expf(st[t][i] - newm[i]);
        st[t][i] = p;
        psum[i] += p;
      }
    }
#pragma unroll
    for (int i = 0; i < 8; ++i) {
      float s = psum[i];
      s += __shfl_xor(s, 1, 32);
      s += __shfl_xor(s, 2, 32);
      s += __shfl_xor(s, 4, 32);
      s += __shfl_xor(s, 8, 32);
      lrow[i] = lrow[i] * alpha[i] + s;
    }
#pragma unroll
    for (int nt = 0; nt < 8; ++nt)
#pragma unroll
      for (int i = 0; i < 8; ++i) o[nt][i] *= alpha[i];

    // P (C-layout) -> wave-private LDS -> re-fragment as A matrix
#pragma unroll
    for (int t = 0; t < 4; ++t)
#pragma unroll
      for (int i = 0; i < 8; ++i)
        pb[(i + 8 * half) * PSTR + t * 16 + n15] = (bf16)st[t][i];

    // O += P * V  (contraction over the 64 kv of this chunk)
#pragma unroll
    for (int kk = 0; kk < CK; kk += 32) {
      v16bf ap = load_a_frag(pb + n15 * PSTR + kk + half * 8);
#pragma unroll
      for (int ng = 0; ng < 2; ++ng) {
        v16bf bfr[4];
#pragma unroll
        for (int nt = 0; nt < 4; ++nt)
          bfr[nt] = load_b_frag(Vc + ((ng * 4 + nt) * 16 + n15) * VSTR + kk + half * 8);
#pragma unroll
        for (int nt = 0; nt < 4; ++nt)
          o[ng * 4 + nt] = __builtin_amdgcn_wmma_f32_16x16x32_bf16(
              false, ap, false, bfr[nt], (short)0, o[ng * 4 + nt], false, false);
      }
    }

#if HAVE_ASYNC_LDS
    if (hasNext) wait_async0();
#else
    if (hasNext) {
      int nb = buf ^ 1;
#pragma unroll
      for (int j = 0; j < 4; ++j) {
        ((uint4*)(Klds + nb * KBUF + kOff))[j] = krg[j];
        ((uint4*)(Vlds + nb * VBUF + vOff))[j] = vrg[j];
      }
    }
#endif
    __syncthreads();
    buf ^= 1;
  }

  // normalize and write attn output [SEQ][NHQ*HD] bf16
#pragma unroll
  for (int i = 0; i < 8; ++i) lrow[i] = 1.0f / lrow[i];
#pragma unroll
  for (int nt = 0; nt < 8; ++nt) {
#pragma unroll
    for (int i = 0; i < 8; ++i) {
      int row = wrow0 + i + 8 * half;
      Om[(size_t)row * (NHQ * HD) + h * HD + nt * 16 + n15] = (bf16)(o[nt][i] * lrow[i]);
    }
  }
}

// ---------------------------------------------------------------------------
// Host-side launcher
// ---------------------------------------------------------------------------
extern "C" void kernel_launch(void* const* d_in, const int* in_sizes, int n_in,
                              void* d_out, int out_size, void* d_ws, size_t ws_size,
                              hipStream_t stream) {
  const float* hs    = (const float*)d_in[0];
  const float* cosb  = (const float*)d_in[1];
  const float* sinb  = (const float*)d_in[2];
  // d_in[3] attention_mask: causal mask applied analytically in-kernel
  const float* pastk = (const float*)d_in[4];
  const float* pastv = (const float*)d_in[5];
  const float* Wq    = (const float*)d_in[6];
  const float* Wk    = (const float*)d_in[7];
  const float* Wv    = (const float*)d_in[8];
  const float* Wo    = (const float*)d_in[9];
  const float* qnw   = (const float*)d_in[10];
  const float* knw   = (const float*)d_in[11];

  float* out_f     = (float*)d_out;                       // [SEQ][HIDN]
  float* new_key   = out_f + (size_t)SEQ * HIDN;          // [NKVH][SEQ][HD]
  float* new_value = new_key + (size_t)NKVH * SEQ * HD;   // [NKVH][SEQ][HD]

  char* ws = (char*)d_ws;
  size_t off = 0;
  auto carve = [&](size_t bytes) {
    void* p = ws + off;
    off += (bytes + 255) & ~(size_t)255;
    return p;
  };
  bf16* hs_b   = (bf16*)carve((size_t)SEQ * HIDN * 2);
  bf16* wq_b   = (bf16*)carve((size_t)NHQ * HD * HIDN * 2);
  bf16* wk_b   = (bf16*)carve((size_t)NKVH * HD * HIDN * 2);
  bf16* wv_b   = (bf16*)carve((size_t)NKVH * HD * HIDN * 2);
  bf16* wo_b   = (bf16*)carve((size_t)HIDN * HIDN * 2);
  float* q_raw = (float*)carve((size_t)SEQ * NHQ * HD * 4);
  float* k_raw = (float*)carve((size_t)SEQ * NKVH * HD * 4);
  float* v_raw = (float*)carve((size_t)SEQ * NKVH * HD * 4);
  bf16* q_b    = (bf16*)carve((size_t)NHQ * SEQ * HD * 2);
  bf16* k_b    = (bf16*)carve((size_t)NKVH * KVTOT * HD * 2);  // row-major K cache
  bf16* vt_b   = (bf16*)carve((size_t)NKVH * HD * KVTOT * 2);  // transposed V cache
  bf16* attn_b = (bf16*)carve((size_t)SEQ * NHQ * HD * 2);

  auto cvt = [&](const float* s, bf16* d, int n) {
    f32_to_bf16_kernel<<<(n + 255) / 256, 256, 0, stream>>>(s, d, n);
  };
  cvt(hs, hs_b, SEQ * HIDN);
  cvt(Wq, wq_b, NHQ * HD * HIDN);
  cvt(Wk, wk_b, NKVH * HD * HIDN);
  cvt(Wv, wv_b, NKVH * HD * HIDN);
  cvt(Wo, wo_b, HIDN * HIDN);
  {
    int n = NKVH * PASTL * HD;
    past_key_kernel<<<(n + 255) / 256, 256, 0, stream>>>(pastk, k_b);
    past_value_kernel<<<(n + 255) / 256, 256, 0, stream>>>(pastv, vt_b);
  }

  // QKV projections (bf16 WMMA, fp32 accumulate)
  gemm_bf16_kernel<<<dim3((NHQ * HD) / GBN, SEQ / GBM), 256, 0, stream>>>(
      hs_b, wq_b, q_raw, SEQ, NHQ * HD, HIDN);
  gemm_bf16_kernel<<<dim3((NKVH * HD) / GBN, SEQ / GBM), 256, 0, stream>>>(
      hs_b, wk_b, k_raw, SEQ, NKVH * HD, HIDN);
  gemm_bf16_kernel<<<dim3((NKVH * HD) / GBN, SEQ / GBM), 256, 0, stream>>>(
      hs_b, wv_b, v_raw, SEQ, NKVH * HD, HIDN);

  // RMSNorm + RoPE: q gets HD^-0.5 folded in; k also feeds new_key (fp32)
  const float qscale = 0.08838834764831845f;  // 128^-0.5
  rmsnorm_rope_kernel<<<SEQ * NHQ, 128, 0, stream>>>(
      q_raw, cosb, sinb, qnw, nullptr, q_b, NHQ, qscale, SEQ, 0);
  rmsnorm_rope_kernel<<<SEQ * NKVH, 128, 0, stream>>>(
      k_raw, cosb, sinb, knw, new_key, k_b, NKVH, 1.0f, KVTOT, PASTL);

  // V passthrough: new_value (fp32) + transposed bf16 cache
  {
    int n = SEQ * NKVH * HD;
    v_finish_kernel<<<(n + 255) / 256, 256, 0, stream>>>(v_raw, new_value, vt_b);
  }

  // Flash attention
  attn_kernel<<<dim3(NHQ, SEQ / ATM), 256, 0, stream>>>(q_b, k_b, vt_b, attn_b);

  // Output projection: out[s,h] = sum_d attn[s,d] * Wo[h,d]
  gemm_bf16_kernel<<<dim3(HIDN / GBN, SEQ / GBM), 256, 0, stream>>>(
      attn_b, wo_b, out_f, SEQ, HIDN, HIDN);
}
